// Retention_76261439308495
// MI455X (gfx1250) — matrix operand order repaired and checked by reading, not compiled, not run on hardware
//
#include <hip/hip_runtime.h>

typedef __attribute__((ext_vector_type(16))) _Float16 v16h;                 // 32B natural
typedef _Float16 v16h_u __attribute__((ext_vector_type(16), aligned(16))); // 16B-aligned view
typedef __attribute__((ext_vector_type(8)))  _Float16 v8h;                  // 16B
typedef __attribute__((ext_vector_type(4)))  _Float16 v4h;                  // 8B
typedef __attribute__((ext_vector_type(8)))  float    v8f;

#define T_SEQ   2048
#define H_HEADS 32
#define D_HEAD  64
#define QSCALE  0.015625f   // 1/64
#define WAVES   8
#define QROWS   (16 * WAVES)  // 128 query rows per block

// ---------------------------------------------------------------------------
// Kernel 1: per-head cumulative sum of log-decay g[B,T,H] -> gc[H][T]
// ---------------------------------------------------------------------------
__global__ void retention_cumsum_kernel(const float* __restrict__ g,
                                        float* __restrict__ gc) {
  int h = threadIdx.x;
  if (h >= H_HEADS) return;
  float run = 0.0f;
  for (int t = 0; t < T_SEQ; ++t) {
    run += g[t * H_HEADS + h];
    gc[h * T_SEQ + t] = run;
  }
}

// ---------------------------------------------------------------------------
// Kernel 2: power retention deg=2, quadratic (attention) form.
// Block = 8 waves (256 thr). Wave w owns query tile t0 = qb*128 + w*16.
// Key tiles processed in pairs (32 keys) so gemm2 uses full K=32 WMMAs.
// All LDS tiles are pre-converted to f16 once per pair (shared by 8 waves).
// ---------------------------------------------------------------------------
__launch_bounds__(256, 2)
__global__ void retention_fwd_kernel(const float* __restrict__ q,
                                     const float* __restrict__ k,
                                     const float* __restrict__ v,
                                     const float* __restrict__ gc,
                                     float* __restrict__ out) {
  // f16 staging; row pads chosen for 16B alignment + bank-conflict-free reads
  __shared__ _Float16 sKh[32][72];        // K rows [s][d], f16
  __shared__ _Float16 sVth[64][40];       // V transposed [d][s], f16
  __shared__ _Float16 sWh[WAVES][16][40]; // per-wave W tile [t_local][s_local]

  const int h    = blockIdx.x % H_HEADS;
  const int qb   = blockIdx.x / H_HEADS;
  const int lane = threadIdx.x & 31;
  const int wave = threadIdx.x >> 5;
  const int t0   = qb * QROWS + wave * 16;
  const int col  = lane & 15;   // N index within 16-lane group
  const int grp  = lane >> 4;   // 0: lanes 0-15, 1: lanes 16-31

  const float* gch = gc + h * T_SEQ;

  // ---- Load Q A-fragments (16x32 f16, two of them for d=0..63), once ----
  // A layout: lanes0-15 elems0-15 -> K=0-15;
  //           lanes16-31 K=16-31 (grp*16).
  v16h a0, a1;
  {
    const float* qr = q + ((size_t)(t0 + col) * H_HEADS + h) * D_HEAD;
#pragma unroll
    for (int e = 0; e < 8; ++e) {
      a0[e]     = (_Float16)qr[grp * 16 + e];
      a0[e + 8] = (_Float16)qr[8 + grp * 16 + e];
      a1[e]     = (_Float16)qr[32 + grp * 16 + e];
      a1[e + 8] = (_Float16)qr[40 + grp * 16 + e];
    }
  }

  // gc for this wave's 8 query rows (C-layout rows: M = r + 8*grp)
  float gct[8];
#pragma unroll
  for (int r = 0; r < 8; ++r) gct[r] = gch[t0 + r + 8 * grp];

  v8f   cn[4] = {v8f{}, v8f{}, v8f{}, v8f{}};  // num accumulators, d tiles 0..3
  float dsum[8] = {0.f, 0.f, 0.f, 0.f, 0.f, 0.f, 0.f, 0.f};

  const int nPairs = (qb + 1) * (QROWS / 32);
  for (int p = 0; p < nPairs; ++p) {
    const int s0 = p * 32;
    __syncthreads();
    // ---- cooperative stage: K -> sKh (f16), V -> sVth (f16, transposed) ----
    {
      const int tid = threadIdx.x;
#pragma unroll
      for (int i = 0; i < 2; ++i) {
        const int f    = tid * 2 + i;        // 0..511 float4 slots
        const int srow = f >> 4;             // 0..31
        const int d4   = (f & 15) * 4;       // 0..60
        const size_t base = ((size_t)(s0 + srow) * H_HEADS + h) * D_HEAD + d4;
        const float4 kf = *(const float4*)(k + base);
        v4h kh;
        kh[0] = (_Float16)kf.x; kh[1] = (_Float16)kf.y;
        kh[2] = (_Float16)kf.z; kh[3] = (_Float16)kf.w;
        *(v4h*)(&sKh[srow][d4]) = kh;
        const float4 vf = *(const float4*)(v + base);
        sVth[d4 + 0][srow] = (_Float16)vf.x;
        sVth[d4 + 1][srow] = (_Float16)vf.y;
        sVth[d4 + 2][srow] = (_Float16)vf.z;
        sVth[d4 + 3][srow] = (_Float16)vf.w;
        if (p + 1 < nPairs) {  // hint next pair into cache hierarchy
          __builtin_prefetch(k + base + 32 * H_HEADS * D_HEAD, 0, 3);
          __builtin_prefetch(v + base + 32 * H_HEADS * D_HEAD, 0, 3);
        }
      }
    }
    __syncthreads();

    // Waves whose whole pair lies above the diagonal contribute nothing.
    if (s0 > t0 + 15) continue;

    // ---- GEMM1: qk tiles for the two 16-wide key tiles of this pair ----
#pragma unroll
    for (int j = 0; j < 2; ++j) {
      const int st = s0 + j * 16;
      // B layout (32x16): lanes0-15 hold K=0..15, lanes16-31 K=16..31
      const _Float16* kr = &sKh[j * 16 + col][0];
      const v16h b0 = (v16h) * (const v16h_u*)(kr + grp * 16);
      const v16h b1 = (v16h) * (const v16h_u*)(kr + 32 + grp * 16);
      v8f c = {};
      c = __builtin_amdgcn_wmma_f32_16x16x32_f16(false, a0, false, b0,
                                                 (short)0, c, false, false);
      c = __builtin_amdgcn_wmma_f32_16x16x32_f16(false, a1, false, b1,
                                                 (short)0, c, false, false);

      // ---- elementwise: w = (qk*SCALE)^2 * exp(gc[t]-gc[s]), causal ----
      const float gcs = gch[st + col];
#pragma unroll
      for (int r = 0; r < 8; ++r) {
        const int tg = t0 + r + 8 * grp;
        const int sg = st + col;
        const float x  = c[r] * QSCALE;
        const float wv = (sg <= tg) ? (x * x * __expf(gct[r] - gcs)) : 0.0f;
        dsum[r] += wv;
        sWh[wave][r + 8 * grp][j * 16 + col] = (_Float16)wv;  // transpose bounce
      }
    }

    // ---- GEMM2: num += W(16x32) x V(32x64) ----
    // Same-wave DS store->load is in-order; read W back in A layout.
    v16h a2;
    {
      const _Float16* wr = &sWh[wave][col][0];
      const v8h lo = *(const v8h*)(wr + grp * 16);       // elems 0..7
      const v8h hi = *(const v8h*)(wr + 8 + grp * 16);   // elems 8..15
#pragma unroll
      for (int e = 0; e < 8; ++e) { a2[e] = lo[e]; a2[e + 8] = hi[e]; }
    }
#pragma unroll
    for (int db = 0; db < 4; ++db) {
      const _Float16* vr = &sVth[db * 16 + col][0];
      const v16h b2 = (v16h) * (const v16h_u*)(vr + grp * 16);
      cn[db] = __builtin_amdgcn_wmma_f32_16x16x32_f16(false, a2, false, b2,
                                                      (short)0, cn[db],
                                                      false, false);
    }
  }

  // ---- denominator: row-sum across the 16 lanes of each group ----
#pragma unroll
  for (int r = 0; r < 8; ++r) {
    float d = dsum[r];
    d += __shfl_xor(d, 1, 32);
    d += __shfl_xor(d, 2, 32);
    d += __shfl_xor(d, 4, 32);
    d += __shfl_xor(d, 8, 32);
    dsum[r] = fmaxf(d, 1.0f);
  }

  // ---- writeback: out[t, h, d] = num / denom ----
#pragma unroll
  for (int db = 0; db < 4; ++db) {
#pragma unroll
    for (int r = 0; r < 8; ++r) {
      const int tg = t0 + r + 8 * grp;
      out[((size_t)tg * H_HEADS + h) * D_HEAD + db * 16 + col] =
          cn[db][r] / dsum[r];
    }
  }
}

// ---------------------------------------------------------------------------
extern "C" void kernel_launch(void* const* d_in, const int* in_sizes, int n_in,
                              void* d_out, int out_size, void* d_ws,
                              size_t ws_size, hipStream_t stream) {
  const float* q = (const float*)d_in[0];
  const float* k = (const float*)d_in[1];
  const float* v = (const float*)d_in[2];
  const float* g = (const float*)d_in[3];
  float* gc = (float*)d_ws;  // H*T floats = 256 KB scratch

  retention_cumsum_kernel<<<1, 32, 0, stream>>>(g, gc);

  const int qblocks = T_SEQ / QROWS;                     // 16
  retention_fwd_kernel<<<dim3(H_HEADS * qblocks), dim3(256), 0, stream>>>(
      q, k, v, gc, (float*)d_out);
}
